// MEFG_42880953483615
// MI455X (gfx1250) — matrix operand order repaired and checked by reading, not compiled, not run on hardware
//
#include <hip/hip_runtime.h>
#include <math.h>

#define BB    8
#define NN    192
#define HID   128
#define MNODE (BB*NN)        // 1536
#define DEG   32
#define ETOT  (MNODE*DEG)    // 49152
#define BN_EPS 1e-5f
#define LN_EPS 1e-5f

typedef __attribute__((ext_vector_type(16))) __bf16 v16bf;
typedef __attribute__((ext_vector_type(8)))  __bf16 v8bf;
typedef __attribute__((ext_vector_type(8)))  float  v8f;

// ---------------------------------------------------------------------------
// WMMA GEMM. 256 threads = 8 waves per block; wave w computes m-tile
// (blockIdx.x*8 + w) x n-tiles [nt0 .. nt0+NPW).  The packed-B fragments for
// the block's whole K-loop are staged ONCE into LDS with CDNA5 async copies
// (global_load_async_to_lds_b128, ASYNCcnt), then all 8 waves read them with
// ds_load_b128.  A is read with global b128 loads (row-major, contiguous K).
// Packed B layout (global): frag[(kt*ntiles + nt)*32 + lane][16] with
//   b[j] = B[kt*32 + (lane>>4)*16 + j, nt*16 + (lane&15)]
// ---------------------------------------------------------------------------
template<int NPW, int BIAS, int ACCUM>
__global__ __launch_bounds__(256)
void gemm_wmma_k(const __bf16* __restrict__ A, long lda, long sAb,
                 const __bf16* __restrict__ Bp, long sBb,
                 const float* __restrict__ bias, float alpha,
                 float* __restrict__ C, long ldc, long sCb,
                 int K, int ntiles, int mtiles)
{
    __shared__ __bf16 Bsh[8 * NPW * 512];          // nk<=8 -> up to 32 KB

    const int tid  = threadIdx.x;
    const int wv   = tid >> 5;
    const int lane = tid & 31;
    const int half = lane >> 4;
    const int r    = lane & 15;
    const int nt0  = blockIdx.y * NPW;
    const int bz   = blockIdx.z;
    A  += (long)bz * sAb;
    Bp += (long)bz * sBb;
    C  += (long)bz * sCb;
    const int nk = K >> 5;

    // ---- stage B fragments (kt in [0,nk), w in [0,NPW)) into LDS, async ----
    // LDS layout: frag (kt*NPW + w) at element offset (kt*NPW+w)*512
    {
        const unsigned lds_base = (unsigned)(unsigned long long)(void*)Bsh;
        for (int it = 0; it < nk; ++it) {
            const int ch   = it * 256 + tid;        // b128 chunk id (8 bf16 each)
            const int frag = ch >> 6;               // 64 chunks per fragment
            const int off  = ch & 63;
            const int kt = frag / NPW, w = frag - kt * NPW;
            const __bf16* g = Bp + ((long)(kt * ntiles + (nt0 + w)) << 9) + off * 8;
            unsigned lds_addr = lds_base + (unsigned)(frag * 512 + off * 8) * 2u;
            unsigned long long ga = (unsigned long long)(const void*)g;
            asm volatile("global_load_async_to_lds_b128 %0, %1, off"
                         :: "v"(lds_addr), "v"(ga) : "memory");
        }
        asm volatile("s_wait_asynccnt 0x0" ::: "memory");
        __syncthreads();
    }

    const int mt = blockIdx.x * 8 + wv;
    if (mt >= mtiles) return;
    const int tm = mt * 16;

    v8f acc[NPW] = {};
    const __bf16* arow = A + (long)(tm + r) * lda;

    for (int kt = 0; kt < nk; ++kt) {
        const int k1 = kt * 32 + half * 16;
        const int k2 = kt * 32 + half * 16 + 8;
        v8bf alo = *(const v8bf*)(arow + k1);
        v8bf ahi = *(const v8bf*)(arow + k2);
        v16bf a = __builtin_shufflevector(alo, ahi,
                    0,1,2,3,4,5,6,7,8,9,10,11,12,13,14,15);
        #pragma unroll
        for (int w = 0; w < NPW; ++w) {
            const __bf16* bp = &Bsh[(kt * NPW + w) * 512 + lane * 16];
            v8bf blo = *(const v8bf*)bp;
            v8bf bhi = *(const v8bf*)(bp + 8);
            v16bf b = __builtin_shufflevector(blo, bhi,
                        0,1,2,3,4,5,6,7,8,9,10,11,12,13,14,15);
            acc[w] = __builtin_amdgcn_wmma_f32_16x16x32_bf16(
                         false, a, false, b, (short)0, acc[w], false, false);
        }
    }

    // C/D layout: VGPR i -> M = i + half*8, N = r
    #pragma unroll
    for (int w = 0; w < NPW; ++w) {
        const int col = (nt0 + w) * 16 + r;
        float bv = BIAS ? bias[col] : 0.f;
        #pragma unroll
        for (int i = 0; i < 8; ++i) {
            const long off = (long)(tm + half * 8 + i) * ldc + col;
            float v = acc[w][i] * alpha + bv;
            if (ACCUM) v += C[off];
            C[off] = v;
        }
    }
}

// pack f32 matrix [K,N] (optionally batched) into WMMA B-fragment layout (bf16)
__global__ void pack_b_k(const float* __restrict__ W, long sInB,
                         __bf16* __restrict__ out, long sOutB,
                         int K, int N, int total)
{
    int idx = blockIdx.x * blockDim.x + threadIdx.x;
    if (idx >= total) return;
    const int per_batch = K * N;
    int b = idx / per_batch;
    int t = idx - b * per_batch;
    int j    = t & 15;
    int lane = (t >> 4) & 31;
    int rem  = t >> 9;            // kt*ntiles + nt
    int ntiles = N >> 4;
    int nt = rem % ntiles;
    int kt = rem / ntiles;
    int n = nt * 16 + (lane & 15);
    int k = kt * 32 + (lane >> 4) * 16 + j;
    out[(long)b * sOutB + t] = (__bf16)W[(long)b * sInB + (long)k * N + n];
}

// y[b, d, i] = bf16(x[b, i, d])   (x: [BB,NN,HID] f32 -> y: [BB,HID,NN] bf16)
__global__ void transpose_bf_k(const float* __restrict__ x, __bf16* __restrict__ y)
{
    int idx = blockIdx.x * blockDim.x + threadIdx.x;
    if (idx >= BB * NN * HID) return;
    int b = idx / (NN * HID);
    int t = idx - b * (NN * HID);
    int d = t / NN;
    int i = t - d * NN;
    y[idx] = (__bf16)x[(long)b * NN * HID + (long)i * HID + d];
}

// ---------------------------------------------------------------------------
// elementwise / reduction helpers
// ---------------------------------------------------------------------------
__global__ void cvt_bf16_k(const float* __restrict__ x, __bf16* __restrict__ y, int n) {
    int i = blockIdx.x * blockDim.x + threadIdx.x;
    if (i < n) y[i] = (__bf16)x[i];
}

__global__ void zero_f32_k(float* __restrict__ x, int n) {
    int i = blockIdx.x * blockDim.x + threadIdx.x;
    if (i < n) x[i] = 0.f;
}

__global__ void conv_edge_k(const float* __restrict__ feat,
                            const int* __restrict__ esrc, const int* __restrict__ edst,
                            const float* __restrict__ cw, const float* __restrict__ cb,
                            __bf16* __restrict__ out)
{
    int idx = blockIdx.x * blockDim.x + threadIdx.x;
    if (idx >= ETOT * HID) return;
    int e = idx >> 7, h = idx & 127;
    const float* fs = feat + (long)esrc[e] * HID;
    const float* fd = feat + (long)edst[e] * HID;
    float acc = cb[0];
    #pragma unroll
    for (int k = 0; k < 3; ++k) {
        int hh = h + k - 1;
        if (hh >= 0 && hh < HID) acc += cw[k] * fs[hh] + cw[3 + k] * fd[hh];
    }
    out[idx] = (__bf16)acc;
}

__global__ void gated_edge_k(const float* __restrict__ Dh, const float* __restrict__ Eh,
                             const float* __restrict__ Bh, const float* __restrict__ e_in,
                             const int* __restrict__ esrc, const int* __restrict__ edst,
                             float* __restrict__ num, float* __restrict__ den)
{
    int idx = blockIdx.x * blockDim.x + threadIdx.x;
    if (idx >= ETOT * HID) return;
    int e = idx >> 7, h = idx & 127;
    int s = esrc[e], d = edst[e];
    float z = Dh[(long)s * HID + h] + Eh[(long)d * HID + h] + e_in[idx];
    float sig = 1.f / (1.f + expf(-z));
    atomicAdd(&num[(long)d * HID + h], Bh[(long)s * HID + h] * sig);
    atomicAdd(&den[(long)d * HID + h], sig);
}

__global__ void node_combine_k(const float* __restrict__ Ah, const float* __restrict__ num,
                               const float* __restrict__ den, float* __restrict__ tmp, int n)
{
    int i = blockIdx.x * blockDim.x + threadIdx.x;
    if (i < n) tmp[i] = Ah[i] + num[i] / (den[i] + 1e-6f);
}

__global__ void bn_stats_k(const float* __restrict__ x, int M,
                           float* __restrict__ mean, float* __restrict__ var)
{
    __shared__ float ss[256], sq[256];
    int c = blockIdx.x, t = threadIdx.x;
    float s = 0.f, q = 0.f;
    for (int r = t; r < M; r += 256) {
        float v = x[(long)r * HID + c];
        s += v; q += v * v;
    }
    ss[t] = s; sq[t] = q; __syncthreads();
    for (int o = 128; o > 0; o >>= 1) {
        if (t < o) { ss[t] += ss[t + o]; sq[t] += sq[t + o]; }
        __syncthreads();
    }
    if (t == 0) {
        float m = ss[0] / (float)M;
        mean[c] = m;
        var[c]  = sq[0] / (float)M - m * m;
    }
}

__global__ void bn_apply_h_k(const float* __restrict__ tmp, const float* __restrict__ feat,
                             const float* __restrict__ mean, const float* __restrict__ var,
                             const float* __restrict__ g, const float* __restrict__ b,
                             float* __restrict__ hs2g, int n)
{
    int i = blockIdx.x * blockDim.x + threadIdx.x;
    if (i >= n) return;
    int c = i & 127;
    float h = (tmp[i] - mean[c]) * rsqrtf(var[c] + BN_EPS) * g[c] + b[c];
    hs2g[i] = feat[i] + fmaxf(h, 0.f);
}

__global__ void bn_apply_out_k(const float* __restrict__ x,
                               const float* __restrict__ mean, const float* __restrict__ var,
                               const float* __restrict__ g, const float* __restrict__ b,
                               float* __restrict__ out, int n)
{
    int i = blockIdx.x * blockDim.x + threadIdx.x;
    if (i >= n) return;
    int c = i & 127;
    float v = (x[i] - mean[c]) * rsqrtf(var[c] + BN_EPS) * g[c] + b[c];
    out[i] = fmaxf(v, 0.f);
}

__global__ void batch_mean_k(const float* __restrict__ hs2g, float* __restrict__ gg)
{
    int b = blockIdx.x, c = threadIdx.x;
    float s = 0.f;
    for (int j = 0; j < NN; ++j) s += hs2g[(long)(b * NN + j) * HID + c];
    gg[b * HID + c] = s / (float)NN;
}

__global__ void enc1_prep_k(const float* __restrict__ gg,
                            const float* __restrict__ Wk, const float* __restrict__ Wv,
                            float* __restrict__ kb, float* __restrict__ vb,
                            float* __restrict__ Ksum)
{
    int b = blockIdx.x, c = threadIdx.x;
    float ak = 0.f, av = 0.f;
    for (int i = 0; i < HID; ++i) {
        float g = gg[b * HID + i];
        ak += g * Wk[i * HID + c];
        av += g * Wv[i * HID + c];
    }
    float kk = (ak > 0.f ? ak : expf(ak) - 1.f) + 1.f;
    kb[b * HID + c]   = kk;
    vb[b * HID + c]   = av;
    Ksum[b * HID + c] = (float)NN * kk;
}

__global__ void outer_kv_k(const float* __restrict__ kb, const float* __restrict__ vb,
                           float* __restrict__ KVf)
{
    int idx = blockIdx.x * blockDim.x + threadIdx.x;
    if (idx >= BB * HID * HID) return;
    int b = idx >> 14, d = (idx >> 7) & 127, e = idx & 127;
    KVf[idx] = kb[b * HID + d] * vb[b * HID + e];
}

__global__ void elu1_dual_k(float* __restrict__ x, __bf16* __restrict__ xb, int n)
{
    int i = blockIdx.x * blockDim.x + threadIdx.x;
    if (i >= n) return;
    float v = x[i];
    v = (v > 0.f ? v : expf(v) - 1.f) + 1.f;
    x[i] = v;
    xb[i] = (__bf16)v;
}

__global__ void elu1_k(float* __restrict__ x, int n)
{
    int i = blockIdx.x * blockDim.x + threadIdx.x;
    if (i >= n) return;
    float v = x[i];
    x[i] = (v > 0.f ? v : expf(v) - 1.f) + 1.f;
}

__global__ void rowdot_scale_k(const float* __restrict__ Q, const float* __restrict__ Ksum,
                               const float* __restrict__ msgmat, float S,
                               __bf16* __restrict__ msg_bf)
{
    __shared__ float red[128];
    int l = blockIdx.x, c = threadIdx.x;
    int b = l / NN;
    float q = Q[(long)l * HID + c];
    red[c] = q * Ksum[b * HID + c];
    __syncthreads();
    for (int o = 64; o > 0; o >>= 1) {
        if (c < o) red[c] += red[c + o];
        __syncthreads();
    }
    float scale = S / (red[0] + 1e-6f);
    msg_bf[(long)l * HID + c] = (__bf16)(msgmat[(long)l * HID + c] * scale);
}

__global__ void ln_k(const float* __restrict__ x, const float* __restrict__ g,
                     const float* __restrict__ b, const float* __restrict__ resid,
                     float* __restrict__ out, __bf16* __restrict__ out_bf)
{
    __shared__ float red[128];
    int l = blockIdx.x, c = threadIdx.x;
    float v = x[(long)l * HID + c];
    red[c] = v; __syncthreads();
    for (int o = 64; o > 0; o >>= 1) { if (c < o) red[c] += red[c + o]; __syncthreads(); }
    float m = red[0] / (float)HID;
    __syncthreads();
    float d = v - m;
    red[c] = d * d; __syncthreads();
    for (int o = 64; o > 0; o >>= 1) { if (c < o) red[c] += red[c + o]; __syncthreads(); }
    float var = red[0] / (float)HID;
    float y = d * rsqrtf(var + LN_EPS) * g[c] + b[c];
    if (resid) y += resid[(long)l * HID + c];
    out[(long)l * HID + c] = y;
    if (out_bf) out_bf[(long)l * HID + c] = (__bf16)y;
}

__global__ void build_cat_k(const float* __restrict__ x, const float* __restrict__ msgLN,
                            __bf16* __restrict__ cat_bf, int n)
{
    int i = blockIdx.x * blockDim.x + threadIdx.x;
    if (i >= n) return;
    int l = i >> 7, c = i & 127;
    cat_bf[(long)l * 256 + c]       = (__bf16)x[i];
    cat_bf[(long)l * 256 + 128 + c] = (__bf16)msgLN[i];
}

__global__ void relu_to_bf_k(const float* __restrict__ x, __bf16* __restrict__ y, int n)
{
    int i = blockIdx.x * blockDim.x + threadIdx.x;
    if (i < n) y[i] = (__bf16)fmaxf(x[i], 0.f);
}

__global__ void colsum_scaled_k(const float* __restrict__ Kq, float* __restrict__ Ksum)
{
    int b = blockIdx.x, c = threadIdx.x;
    float s = 0.f;
    for (int i = 0; i < NN; ++i) s += Kq[(long)(b * NN + i) * HID + c];
    Ksum[b * HID + c] = (float)NN * s;
}

__global__ void gather_ge_k(const float* __restrict__ enc2out,
                            const int* __restrict__ esrc, __bf16* __restrict__ ge_bf)
{
    int idx = blockIdx.x * blockDim.x + threadIdx.x;
    if (idx >= ETOT * HID) return;
    int e = idx >> 7, c = idx & 127;
    ge_bf[idx] = (__bf16)enc2out[(long)esrc[e] * HID + c];
}

// ---------------------------------------------------------------------------
// host orchestration
// ---------------------------------------------------------------------------
static inline void* bump(char*& p, size_t bytes) {
    void* r = (void*)p;
    p += (bytes + 255) & ~(size_t)255;
    return r;
}

static inline void launch_gemm(hipStream_t s,
    const __bf16* A, long lda, long sAb,
    const __bf16* Bp, long sBb,
    const float* bias, float alpha, int accum,
    float* C, long ldc, long sCb, int K, int Nc, int mtiles, int batches)
{
    const int ntiles = Nc / 16;
    dim3 g((mtiles + 7) / 8, ntiles / 4, batches);
    if (accum)
        gemm_wmma_k<4,1,1><<<g, 256, 0, s>>>(A, lda, sAb, Bp, sBb, bias, alpha, C, ldc, sCb, K, ntiles, mtiles);
    else if (bias)
        gemm_wmma_k<4,1,0><<<g, 256, 0, s>>>(A, lda, sAb, Bp, sBb, bias, alpha, C, ldc, sCb, K, ntiles, mtiles);
    else
        gemm_wmma_k<4,0,0><<<g, 256, 0, s>>>(A, lda, sAb, Bp, sBb, bias, alpha, C, ldc, sCb, K, ntiles, mtiles);
}

static inline void launch_pack(hipStream_t s, const float* W, long sInB,
                               __bf16* out, int K, int N, int batches)
{
    int total = batches * K * N;
    pack_b_k<<<(total + 255) / 256, 256, 0, s>>>(W, sInB, out, (long)K * N, K, N, total);
}

struct EncBufs {
    float  *Qf, *msgmat, *msgWm, *h1, *h2;
    __bf16 *Q_bf, *msg_bf, *cat_bf, *h1_bf;
};

static void run_encoder(hipStream_t stream,
    const float* x, const __bf16* x_bf,
    const __bf16* KVp, const float* Ksum, float S,
    const __bf16* Wqp, const __bf16* Wmp,
    const __bf16* Wm1p, const __bf16* Wm2p,
    const float* ln1_g, const float* ln1_b,
    const float* ln2_g, const float* ln2_b,
    EncBufs& t, float* out, __bf16* out_bf)
{
    const int n128 = MNODE * HID;
    // Q = elu(x@Wq)+1
    launch_gemm(stream, x_bf, HID, 0, Wqp, 0, nullptr, 1.f, 0,
                t.Qf, HID, 0, HID, HID, MNODE / 16, 1);
    elu1_dual_k<<<(n128 + 255) / 256, 256, 0, stream>>>(t.Qf, t.Q_bf, n128);
    // msgmat[b] = Q[b] @ KV[b]
    launch_gemm(stream, t.Q_bf, HID, (long)NN * HID, KVp, (long)HID * HID,
                nullptr, 1.f, 0, t.msgmat, HID, (long)NN * HID, HID, HID, NN / 16, BB);
    rowdot_scale_k<<<MNODE, 128, 0, stream>>>(t.Qf, Ksum, t.msgmat, S, t.msg_bf);
    // msg = LN(msg @ Wm)
    launch_gemm(stream, t.msg_bf, HID, 0, Wmp, 0, nullptr, 1.f, 0,
                t.msgWm, HID, 0, HID, HID, MNODE / 16, 1);
    ln_k<<<MNODE, 128, 0, stream>>>(t.msgWm, ln1_g, ln1_b, nullptr, t.msgWm, nullptr);
    // h2 = relu([x,msg]@Wm1)@Wm2 ; out = x + LN(h2)
    build_cat_k<<<(n128 + 255) / 256, 256, 0, stream>>>(x, t.msgWm, t.cat_bf, n128);
    launch_gemm(stream, t.cat_bf, 256, 0, Wm1p, 0, nullptr, 1.f, 0,
                t.h1, 256, 0, 256, 256, MNODE / 16, 1);
    relu_to_bf_k<<<(MNODE * 256 + 255) / 256, 256, 0, stream>>>(t.h1, t.h1_bf, MNODE * 256);
    launch_gemm(stream, t.h1_bf, 256, 0, Wm2p, 0, nullptr, 1.f, 0,
                t.h2, HID, 0, 256, HID, MNODE / 16, 1);
    ln_k<<<MNODE, 128, 0, stream>>>(t.h2, ln2_g, ln2_b, x, out, out_bf);
}

extern "C" void kernel_launch(void* const* d_in, const int* in_sizes, int n_in,
                              void* d_out, int out_size, void* d_ws, size_t ws_size,
                              hipStream_t stream)
{
    (void)in_sizes; (void)n_in; (void)out_size; (void)ws_size;
    const float* feat   = (const float*)d_in[0];
    const float* e_in   = (const float*)d_in[1];
    const int*   esrc   = (const int*)d_in[2];
    const int*   edst   = (const int*)d_in[3];
    const float* conv_w = (const float*)d_in[4];
    const float* conv_b = (const float*)d_in[5];
    const float* W2 = (const float*)d_in[6],  *b2 = (const float*)d_in[7];
    const float* W3 = (const float*)d_in[8],  *b3 = (const float*)d_in[9];
    const float* W4 = (const float*)d_in[10], *b4 = (const float*)d_in[11];
    const float* bn_lr_g = (const float*)d_in[12], *bn_lr_b = (const float*)d_in[13];
    const float* WA = (const float*)d_in[14], *bA = (const float*)d_in[15];
    const float* WBm= (const float*)d_in[16], *bBv= (const float*)d_in[17];
    const float* WD = (const float*)d_in[18], *bD = (const float*)d_in[19];
    const float* WE = (const float*)d_in[20], *bE = (const float*)d_in[21];
    const float* bnh_g = (const float*)d_in[22], *bnh_b = (const float*)d_in[23];
    const float* Wq = (const float*)d_in[24], *Wk = (const float*)d_in[25];
    const float* Wv = (const float*)d_in[26], *Wm = (const float*)d_in[27];
    const float* Wm1 = (const float*)d_in[28], *Wm2 = (const float*)d_in[29];
    const float* ln1_g = (const float*)d_in[30], *ln1_b = (const float*)d_in[31];
    const float* ln2_g = (const float*)d_in[32], *ln2_b = (const float*)d_in[33];

    const int n128 = MNODE * HID;     // 196608
    const int nE   = ETOT * HID;      // 6291456
    char* ws = (char*)d_ws;

    __bf16* feat_bf = (__bf16*)bump(ws, (size_t)n128 * 2);
    __bf16 *W2p, *W3p, *W4p, *WAp, *WBp, *WDp, *WEp, *Wqp, *Wkp, *Wvp, *Wmp;
    const float* wsrc[11] = { W2, W3, W4, WA, WBm, WD, WE, Wq, Wk, Wv, Wm };
    __bf16** wdst[11]     = { &W2p, &W3p, &W4p, &WAp, &WBp, &WDp, &WEp, &Wqp, &Wkp, &Wvp, &Wmp };
    for (int i = 0; i < 11; ++i) *wdst[i] = (__bf16*)bump(ws, (size_t)HID * HID * 2);
    __bf16* Wm1p = (__bf16*)bump(ws, (size_t)256 * 256 * 2);
    __bf16* Wm2p = (__bf16*)bump(ws, (size_t)256 * HID * 2);

    __bf16* convge_bf = (__bf16*)bump(ws, (size_t)nE * 2);   // conv operand, later ge operand
    float*  lr_local  = (float*)bump(ws, (size_t)nE * 4);

    float *hs  = (float*)bump(ws, (size_t)n128 * 4);
    float *Ah  = (float*)bump(ws, (size_t)n128 * 4);
    float *Bh  = (float*)bump(ws, (size_t)n128 * 4);
    float *Dh  = (float*)bump(ws, (size_t)n128 * 4);
    float *Eh  = (float*)bump(ws, (size_t)n128 * 4);
    float *num = (float*)bump(ws, (size_t)n128 * 4);
    float *den = (float*)bump(ws, (size_t)n128 * 4);
    float *tmp = (float*)bump(ws, (size_t)n128 * 4);
    float *mean = (float*)bump(ws, 512);
    float *var  = (float*)bump(ws, 512);
    float *hs2g = (float*)bump(ws, (size_t)n128 * 4);
    float *gg   = (float*)bump(ws, (size_t)BB * HID * 4);
    __bf16* hs_bf = (__bf16*)bump(ws, (size_t)n128 * 2);

    EncBufs t;
    t.Qf     = (float*)bump(ws, (size_t)n128 * 4);
    t.msgmat = (float*)bump(ws, (size_t)n128 * 4);
    t.msgWm  = (float*)bump(ws, (size_t)n128 * 4);
    t.h1     = (float*)bump(ws, (size_t)MNODE * 256 * 4);
    t.h2     = (float*)bump(ws, (size_t)n128 * 4);
    t.Q_bf   = (__bf16*)bump(ws, (size_t)n128 * 2);
    t.msg_bf = (__bf16*)bump(ws, (size_t)n128 * 2);
    t.cat_bf = (__bf16*)bump(ws, (size_t)MNODE * 256 * 2);
    t.h1_bf  = (__bf16*)bump(ws, (size_t)MNODE * 256 * 2);

    float  *KVf  = (float*)bump(ws, (size_t)BB * HID * HID * 4);
    __bf16 *KVp  = (__bf16*)bump(ws, (size_t)BB * HID * HID * 2);
    float  *Ksum = (float*)bump(ws, (size_t)BB * HID * 4);
    float  *kb   = (float*)bump(ws, (size_t)BB * HID * 4);
    float  *vb   = (float*)bump(ws, (size_t)BB * HID * 4);

    float  *q     = (float*)bump(ws, (size_t)n128 * 4);
    __bf16 *q_bf  = (__bf16*)bump(ws, (size_t)n128 * 2);
    float  *Kqf   = (float*)bump(ws, (size_t)n128 * 4);
    __bf16 *Kq_t  = (__bf16*)bump(ws, (size_t)n128 * 2);   // [BB,HID,NN] bf16
    float  *Vqf   = (float*)bump(ws, (size_t)n128 * 4);
    __bf16 *Vqp   = (__bf16*)bump(ws, (size_t)n128 * 2);   // packed per batch
    float  *enc2out = (float*)bump(ws, (size_t)n128 * 4);

    // --- operand prep ---
    cvt_bf16_k<<<(n128 + 255) / 256, 256, 0, stream>>>(feat, feat_bf, n128);
    for (int i = 0; i < 11; ++i) launch_pack(stream, wsrc[i], 0, *wdst[i], HID, HID, 1);
    launch_pack(stream, Wm1, 0, Wm1p, 256, 256, 1);
    launch_pack(stream, Wm2, 0, Wm2p, 256, HID, 1);

    // --- node linears ---
    const int MT = MNODE / 16;
    launch_gemm(stream, feat_bf, HID, 0, W3p, 0, b3, 1.f, 0, hs, HID, 0, HID, HID, MT, 1);
    launch_gemm(stream, feat_bf, HID, 0, WAp, 0, bA, 1.f, 0, Ah, HID, 0, HID, HID, MT, 1);
    launch_gemm(stream, feat_bf, HID, 0, WBp, 0, bBv,1.f, 0, Bh, HID, 0, HID, HID, MT, 1);
    launch_gemm(stream, feat_bf, HID, 0, WDp, 0, bD, 1.f, 0, Dh, HID, 0, HID, HID, MT, 1);
    launch_gemm(stream, feat_bf, HID, 0, WEp, 0, bE, 1.f, 0, Eh, HID, 0, HID, HID, MT, 1);

    // --- local edge path ---
    conv_edge_k<<<(nE + 255) / 256, 256, 0, stream>>>(feat, esrc, edst, conv_w, conv_b, convge_bf);
    launch_gemm(stream, convge_bf, HID, 0, W2p, 0, b2, 1.f, 0,
                lr_local, HID, 0, HID, HID, ETOT / 16, 1);

    // --- GatedGCN ---
    zero_f32_k<<<(n128 + 255) / 256, 256, 0, stream>>>(num, n128);
    zero_f32_k<<<(n128 + 255) / 256, 256, 0, stream>>>(den, n128);
    gated_edge_k<<<(nE + 255) / 256, 256, 0, stream>>>(Dh, Eh, Bh, e_in, esrc, edst, num, den);
    node_combine_k<<<(n128 + 255) / 256, 256, 0, stream>>>(Ah, num, den, tmp, n128);
    bn_stats_k<<<HID, 256, 0, stream>>>(tmp, MNODE, mean, var);
    bn_apply_h_k<<<(n128 + 255) / 256, 256, 0, stream>>>(tmp, feat, mean, var, bnh_g, bnh_b, hs2g, n128);
    batch_mean_k<<<BB, 128, 0, stream>>>(hs2g, gg);

    // --- encoder 1 (rank-1 KV) ---
    cvt_bf16_k<<<(n128 + 255) / 256, 256, 0, stream>>>(hs, hs_bf, n128);
    enc1_prep_k<<<BB, 128, 0, stream>>>(gg, Wk, Wv, kb, vb, Ksum);
    outer_kv_k<<<(BB * HID * HID + 255) / 256, 256, 0, stream>>>(kb, vb, KVf);
    launch_pack(stream, KVf, (long)HID * HID, KVp, HID, HID, BB);
    run_encoder(stream, hs, hs_bf, KVp, Ksum, (float)NN,
                Wqp, Wmp, Wm1p, Wm2p, ln1_g, ln1_b, ln2_g, ln2_b, t, q, q_bf);

    // --- encoder 2 (output independent of "end" index) ---
    launch_gemm(stream, q_bf, HID, 0, Wkp, 0, nullptr, 1.f, 0, Kqf, HID, 0, HID, HID, MT, 1);
    elu1_k<<<(n128 + 255) / 256, 256, 0, stream>>>(Kqf, n128);
    launch_gemm(stream, q_bf, HID, 0, Wvp, 0, nullptr, 1.f, 0, Vqf, HID, 0, HID, HID, MT, 1);
    transpose_bf_k<<<(n128 + 255) / 256, 256, 0, stream>>>(Kqf, Kq_t);
    launch_pack(stream, Vqf, (long)NN * HID, Vqp, NN, HID, BB);
    // KV[b] = (1/N) Kq[b]^T @ Vq[b]
    launch_gemm(stream, Kq_t, NN, (long)HID * NN, Vqp, (long)NN * HID,
                nullptr, 1.f / (float)NN, 0, KVf, HID, (long)HID * HID, NN, HID, HID / 16, BB);
    launch_pack(stream, KVf, (long)HID * HID, KVp, HID, HID, BB);
    colsum_scaled_k<<<BB, 128, 0, stream>>>(Kqf, Ksum);
    run_encoder(stream, q, q_bf, KVp, Ksum, (float)(NN * NN),
                Wqp, Wmp, Wm1p, Wm2p, ln1_g, ln1_b, ln2_g, ln2_b, t, enc2out, nullptr);

    // --- fuse + final bn/relu ---
    gather_ge_k<<<(nE + 255) / 256, 256, 0, stream>>>(enc2out, esrc, convge_bf);
    launch_gemm(stream, convge_bf, HID, 0, W4p, 0, b4, 1.f, 1,
                lr_local, HID, 0, HID, HID, ETOT / 16, 1);
    bn_stats_k<<<HID, 256, 0, stream>>>(lr_local, ETOT, mean, var);
    bn_apply_out_k<<<(nE + 255) / 256, 256, 0, stream>>>(lr_local, mean, var, bn_lr_g, bn_lr_b,
                                                         (float*)d_out, nE);
}
